// SwinTransBlock_1580547964956
// MI455X (gfx1250) — matrix operand order repaired and checked by reading, not compiled, not run on hardware
//
#include <hip/hip_runtime.h>

typedef __attribute__((ext_vector_type(16))) _Float16 v16h;
typedef __attribute__((ext_vector_type(8)))  float    v8f;

#define DEVINL __device__ __forceinline__

static constexpr int SS    = 4;
static constexpr int Cc    = 256;
static constexpr int TOK   = 8 * 64 * 64;   // 32768 tokens
static constexpr int WINS  = 512;           // B * nW
static constexpr float QSCALE = 0.17677669529663687f; // 32^-0.5

DEVINL v8f zero8() {
  v8f z;
#pragma unroll
  for (int i = 0; i < 8; ++i) z[i] = 0.f;
  return z;
}

DEVINL v8f wmma16(v16h a, v16h b, v8f c) {
  // v_wmma_f32_16x16x32_f16  (D = A*B + C, f32 accum)
  return __builtin_amdgcn_wmma_f32_16x16x32_f16(false, a, false, b, (short)0, c,
                                                false, false);
}

// Async memory->LDS 128-bit copy (VGLOBAL GLOBAL_LOAD_ASYNC_TO_LDS_B128,
// tracked by ASYNCcnt). VDST = LDS byte address (low 32 bits of the generic
// pointer == AS3 offset per ISA flat->LDS mapping), VADDR = 64-bit global.
DEVINL void async_copy128(const void* g, void* l) {
  asm volatile("global_load_async_to_lds_b128 %0, %1, off"
               :
               : "v"((unsigned)(size_t)l), "v"((unsigned long long)(size_t)g)
               : "memory");
}
DEVINL void wait_async0() {
  asm volatile("s_wait_asynccnt 0x0" ::: "memory");
}

// A fragment: 16x32 f16, row-major source. Lane L holds row L%16;
// element e -> K = (L/16)*16 + e
DEVINL v16h load_a(const _Float16* A, int lda, int row0, int k0, int lane) {
  const _Float16* p = A + (size_t)(row0 + (lane & 15)) * lda + k0 + ((lane >> 4) << 4);
  v16h a;
#pragma unroll
  for (int e = 0; e < 16; ++e) a[e] = p[e];
  return a;
}

// B = W^T fragment where W is row-major [n][k]: lane holds column n0+L%16,
// element e -> K = (L/16)*16 + e  (contiguous 32B per lane)
DEVINL v16h load_bt(const _Float16* W, int ldw, int n0, int k0, int lane) {
  const _Float16* p = W + (size_t)(n0 + (lane & 15)) * ldw + k0 + ((lane >> 4) << 4);
  v16h b;
#pragma unroll
  for (int e = 0; e < 16; ++e) b[e] = p[e];
  return b;
}

// B fragment from a row-major [k][n] matrix
DEVINL v16h load_b(const _Float16* Bm, int ldb, int n0, int k0, int lane) {
  const _Float16* p = Bm + (size_t)(k0 + ((lane >> 4) << 4)) * ldb + n0 + (lane & 15);
  v16h b;
#pragma unroll
  for (int e = 0; e < 16; ++e) b[e] = p[(size_t)e * ldb];
  return b;
}

// ---------------------------------------------------------------- convert
__global__ void k_f32_to_f16(const float* __restrict__ s, _Float16* __restrict__ d, int n) {
  int i = blockIdx.x * blockDim.x + threadIdx.x;
  if (i < n) d[i] = (_Float16)s[i];
}

// ------------------------------------------- LN1 + shift + window partition
__global__ void k_ln1_shift(const float* __restrict__ x, const float* __restrict__ g,
                            const float* __restrict__ be, _Float16* __restrict__ hwin) {
  int lane = threadIdx.x & 31, wave = threadIdx.x >> 5;
  int t = blockIdx.x * 8 + wave;                       // one wave per token
  const float* xt = x + (size_t)t * Cc;
  float v[8]; float s = 0.f;
#pragma unroll
  for (int u = 0; u < 8; ++u) { v[u] = xt[lane + 32 * u]; s += v[u]; }
#pragma unroll
  for (int off = 16; off; off >>= 1) s += __shfl_xor(s, off, 32);
  float mu = s * (1.f / 256.f);
  float q = 0.f;
#pragma unroll
  for (int u = 0; u < 8; ++u) { float d = v[u] - mu; q += d * d; }
#pragma unroll
  for (int off = 16; off; off >>= 1) q += __shfl_xor(q, off, 32);
  float rs = rsqrtf(q * (1.f / 256.f) + 1e-5f);
  int b = t >> 12, idx = t & 4095, r = idx >> 6, c = idx & 63;
  int r2 = (r - SS) & 63, c2 = (c - SS) & 63;          // cyclic shift
  int wl = ((r2 >> 3) << 3) + (c2 >> 3);               // window in image
  int n  = ((r2 & 7) << 3) + (c2 & 7);                 // token in window
  _Float16* dst = hwin + (size_t)((b * 64 + wl) * 64 + n) * Cc;
#pragma unroll
  for (int u = 0; u < 8; ++u) {
    int ch = lane + 32 * u;
    dst[ch] = (_Float16)((v[u] - mu) * rs * g[ch] + be[ch]);
  }
}

// ---------------------------------------------------------------- QKV GEMM
__global__ void k_qkv(const _Float16* __restrict__ hwin, const _Float16* __restrict__ Wq,
                      const float* __restrict__ bq, _Float16* __restrict__ qo,
                      _Float16* __restrict__ ko, _Float16* __restrict__ vo) {
  __shared__ _Float16 As[64 * 256];
  int win = blockIdx.x, tid = threadIdx.x, lane = tid & 31, wave = tid >> 5;
  {
    const char* src = (const char*)(hwin + (size_t)win * 64 * 256);
    char* dst = (char*)As;
#pragma unroll
    for (int u = 0; u < 8; ++u) {
      int bo = (tid + 256 * u) * 16;
      async_copy128(src + bo, dst + bo);               // ASYNCcnt-tracked DMA
    }
    wait_async0();
  }
  __syncthreads();
  for (int tix = wave; tix < 192; tix += 8) {          // 4 row-tiles x 48 col-tiles
    int row0 = (tix & 3) * 16, n0 = (tix >> 2) * 16;
    v8f acc = zero8();
#pragma unroll
    for (int kk = 0; kk < 8; ++kk) {
      v16h a = load_a(As, 256, row0, kk * 32, lane);
      v16h b = load_bt(Wq, 256, n0, kk * 32, lane);
      acc = wmma16(a, b, acc);
    }
    int n = n0 + (lane & 15);
    int which = n >> 8, cc = n & 255, head = cc >> 5, hd = cc & 31;
    _Float16* outp = (which == 0) ? qo : (which == 1) ? ko : vo;
    float bias = bq[n];
    float scl = (which == 0) ? QSCALE : 1.f;           // fold softmax scale into Q
#pragma unroll
    for (int r = 0; r < 8; ++r) {
      int tok = row0 + r + ((lane >> 4) << 3);
      outp[(size_t)((win * 8 + head) * 64 + tok) * 32 + hd] =
          (_Float16)((acc[r] + bias) * scl);
    }
  }
}

// --------------------------------------------------------------- attention
__global__ void k_attn(const _Float16* __restrict__ q, const _Float16* __restrict__ km,
                       const _Float16* __restrict__ vm, const float* __restrict__ rpb,
                       _Float16* __restrict__ aout) {
  __shared__ float sbias[225 * 8];
  extern __shared__ _Float16 Pl[];                     // [8 heads][64*72] f16
  int win = blockIdx.x, tid = threadIdx.x, lane = tid & 31, head = tid >> 5;
  for (int i = tid; i < 225 * 8; i += 256) sbias[i] = rpb[i];
  __syncthreads();
  const _Float16* qh = q  + (size_t)(win * 8 + head) * 64 * 32;
  const _Float16* kh = km + (size_t)(win * 8 + head) * 64 * 32;
  const _Float16* vh = vm + (size_t)(win * 8 + head) * 64 * 32;
  _Float16* P = Pl + head * (64 * 72);
  int wl = win & 63, wr0 = (wl >> 3) << 3, wc0 = (wl & 7) << 3;
  int lh = lane >> 4;
  for (int i = 0; i < 4; ++i) {                        // S = Q K^T, row-block at a time
    v16h a = load_a(qh, 32, i * 16, 0, lane);
    v8f s[4];
#pragma unroll
    for (int j = 0; j < 4; ++j) {
      v16h b = load_bt(kh, 32, j * 16, 0, lane);       // K^T: one WMMA per 16x16 tile
      s[j] = wmma16(a, b, zero8());
    }
#pragma unroll
    for (int j = 0; j < 4; ++j) {                      // rel-pos bias + shift mask
      int m = j * 16 + (lane & 15);
      int mr = m >> 3, mc = m & 7, gmr = wr0 + mr, gmc = wc0 + mc;
      int idm = (gmr < 56 ? 0 : gmr < 60 ? 1 : 2) * 3 + (gmc < 56 ? 0 : gmc < 60 ? 1 : 2);
#pragma unroll
      for (int r = 0; r < 8; ++r) {
        int nq = i * 16 + r + (lh << 3);
        int nr = nq >> 3, nc = nq & 7, gnr = wr0 + nr, gnc = wc0 + nc;
        int idn = (gnr < 56 ? 0 : gnr < 60 ? 1 : 2) * 3 + (gnc < 56 ? 0 : gnc < 60 ? 1 : 2);
        int ridx = (nr - mr + 7) * 15 + (nc - mc + 7);
        s[j][r] += sbias[ridx * 8 + head] + (idn != idm ? -100.f : 0.f);
      }
    }
    float rmax[8], rsum[8];
#pragma unroll
    for (int r = 0; r < 8; ++r) {                      // row max (16-lane halves)
      float m0 = fmaxf(fmaxf(s[0][r], s[1][r]), fmaxf(s[2][r], s[3][r]));
#pragma unroll
      for (int off = 8; off; off >>= 1) m0 = fmaxf(m0, __shfl_xor(m0, off, 32));
      rmax[r] = m0;
    }
#pragma unroll
    for (int j = 0; j < 4; ++j)
#pragma unroll
      for (int r = 0; r < 8; ++r) s[j][r] = __expf(s[j][r] - rmax[r]);
#pragma unroll
    for (int r = 0; r < 8; ++r) {
      float t = s[0][r] + s[1][r] + s[2][r] + s[3][r];
#pragma unroll
      for (int off = 8; off; off >>= 1) t += __shfl_xor(t, off, 32);
      rsum[r] = 1.f / t;
    }
#pragma unroll
    for (int j = 0; j < 4; ++j) {                      // P -> LDS (f16, padded)
      int col = j * 16 + (lane & 15);
#pragma unroll
      for (int r = 0; r < 8; ++r)
        P[(i * 16 + r + (lh << 3)) * 72 + col] = (_Float16)(s[j][r] * rsum[r]);
    }
  }
  // O = P @ V  (same wave: DScnt ordering handled by compiler)
#pragma unroll
  for (int i = 0; i < 4; ++i) {
#pragma unroll
    for (int j2 = 0; j2 < 2; ++j2) {
      v8f acc = zero8();
#pragma unroll
      for (int kk = 0; kk < 2; ++kk) {
        v16h a = load_a(P, 72, i * 16, kk * 32, lane);
        v16h b = load_b(vh, 32, j2 * 16, kk * 32, lane);
        acc = wmma16(a, b, acc);
      }
      int ch = head * 32 + j2 * 16 + (lane & 15);
#pragma unroll
      for (int r = 0; r < 8; ++r) {
        int tok = i * 16 + r + (lh << 3);
        aout[(size_t)(win * 64 + tok) * 256 + ch] = (_Float16)acc[r];
      }
    }
  }
}

// ------------------------- proj GEMM + window reverse + unshift + residual
__global__ void k_proj(const _Float16* __restrict__ aout, const _Float16* __restrict__ Wp,
                       const float* __restrict__ bp, const float* __restrict__ x,
                       float* __restrict__ x1) {
  __shared__ _Float16 As[64 * 256];
  int win = blockIdx.x, tid = threadIdx.x, lane = tid & 31, wave = tid >> 5;
  {
    const char* src = (const char*)(aout + (size_t)win * 64 * 256);
    char* dst = (char*)As;
#pragma unroll
    for (int u = 0; u < 8; ++u) {
      int bo = (tid + 256 * u) * 16;
      async_copy128(src + bo, dst + bo);
    }
    wait_async0();
  }
  __syncthreads();
  int b = win >> 6, wl = win & 63;
  for (int tix = wave; tix < 64; tix += 8) {           // 4 x 16 tiles
    int row0 = (tix & 3) * 16, n0 = (tix >> 2) * 16;
    v8f acc = zero8();
#pragma unroll
    for (int kk = 0; kk < 8; ++kk) {
      v16h a = load_a(As, 256, row0, kk * 32, lane);
      v16h bb = load_bt(Wp, 256, n0, kk * 32, lane);
      acc = wmma16(a, bb, acc);
    }
    int col = n0 + (lane & 15);
    float bias = bp[col];
#pragma unroll
    for (int r = 0; r < 8; ++r) {
      int n = row0 + r + ((lane >> 4) << 3);
      int r2 = ((wl >> 3) << 3) + (n >> 3), c2 = ((wl & 7) << 3) + (n & 7);
      int rr = (r2 + SS) & 63, cc2 = (c2 + SS) & 63;   // reverse cyclic shift
      size_t t = (size_t)(b * 4096 + rr * 64 + cc2);
      x1[t * 256 + col] = x[t * 256 + col] + acc[r] + bias;
    }
  }
}

// --------------------------------------------------------------------- LN2
__global__ void k_ln2(const float* __restrict__ x1, const float* __restrict__ g,
                      const float* __restrict__ be, _Float16* __restrict__ h2) {
  int lane = threadIdx.x & 31, wave = threadIdx.x >> 5;
  int t = blockIdx.x * 8 + wave;
  const float* xt = x1 + (size_t)t * Cc;
  float v[8]; float s = 0.f;
#pragma unroll
  for (int u = 0; u < 8; ++u) { v[u] = xt[lane + 32 * u]; s += v[u]; }
#pragma unroll
  for (int off = 16; off; off >>= 1) s += __shfl_xor(s, off, 32);
  float mu = s * (1.f / 256.f);
  float q = 0.f;
#pragma unroll
  for (int u = 0; u < 8; ++u) { float d = v[u] - mu; q += d * d; }
#pragma unroll
  for (int off = 16; off; off >>= 1) q += __shfl_xor(q, off, 32);
  float rs = rsqrtf(q * (1.f / 256.f) + 1e-5f);
  _Float16* dst = h2 + (size_t)t * Cc;
#pragma unroll
  for (int u = 0; u < 8; ++u) {
    int ch = lane + 32 * u;
    dst[ch] = (_Float16)((v[u] - mu) * rs * g[ch] + be[ch]);
  }
}

// ---------------------------------------------------------- FC1 + GELU
__global__ void k_fc1(const _Float16* __restrict__ h2, const _Float16* __restrict__ W1,
                      const float* __restrict__ b1v, _Float16* __restrict__ h3) {
  __shared__ _Float16 As[64 * 256];
  int rb = blockIdx.x, cb = blockIdx.y;
  int tid = threadIdx.x, lane = tid & 31, wave = tid >> 5;
  {
    const char* src = (const char*)(h2 + (size_t)rb * 64 * 256);
    char* dst = (char*)As;
#pragma unroll
    for (int u = 0; u < 8; ++u) {
      int bo = (tid + 256 * u) * 16;
      async_copy128(src + bo, dst + bo);
    }
    wait_async0();
  }
  __syncthreads();
#pragma unroll
  for (int t2 = 0; t2 < 2; ++t2) {                     // 16 tiles / 8 waves
    int tix = wave + t2 * 8;
    int row0 = (tix & 3) * 16, n0 = cb * 64 + (tix >> 2) * 16;
    v8f acc = zero8();
#pragma unroll
    for (int kk = 0; kk < 8; ++kk) {
      v16h a = load_a(As, 256, row0, kk * 32, lane);
      v16h bb = load_bt(W1, 256, n0, kk * 32, lane);
      acc = wmma16(a, bb, acc);
    }
    int col = n0 + (lane & 15);
    float bias = b1v[col];
#pragma unroll
    for (int r = 0; r < 8; ++r) {
      size_t row = (size_t)rb * 64 + row0 + r + ((lane >> 4) << 3);
      float y = acc[r] + bias;
      y = 0.5f * y * (1.f + erff(y * 0.70710678118654752f));   // exact GELU
      h3[row * 1024 + col] = (_Float16)y;
    }
  }
}

// ----------------------------------------------------- FC2 + residual out
__global__ void k_fc2(const _Float16* __restrict__ h3, const _Float16* __restrict__ W2,
                      const float* __restrict__ b2v, const float* __restrict__ x1,
                      float* __restrict__ out) {
  __shared__ _Float16 As[64 * 256];
  int rb = blockIdx.x, cb = blockIdx.y;
  int tid = threadIdx.x, lane = tid & 31, wave = tid >> 5;
  v8f acc[2]; acc[0] = zero8(); acc[1] = zero8();
  for (int kc = 0; kc < 4; ++kc) {                     // K=1024 in 256-chunks via LDS
    __syncthreads();
#pragma unroll
    for (int u = 0; u < 8; ++u) {
      int uu = tid + 256 * u;                          // 2048 x 16B async moves
      int row = uu >> 5, off = (uu & 31) << 3;
      async_copy128(h3 + (size_t)(rb * 64 + row) * 1024 + kc * 256 + off,
                    As + row * 256 + off);
    }
    wait_async0();
    __syncthreads();
#pragma unroll
    for (int t2 = 0; t2 < 2; ++t2) {
      int tix = wave + t2 * 8;
      int row0 = (tix & 3) * 16, n0 = cb * 64 + (tix >> 2) * 16;
#pragma unroll
      for (int kk = 0; kk < 8; ++kk) {
        v16h a = load_a(As, 256, row0, kk * 32, lane);
        v16h bb = load_bt(W2, 1024, n0, kc * 256 + kk * 32, lane);
        acc[t2] = wmma16(a, bb, acc[t2]);
      }
    }
  }
#pragma unroll
  for (int t2 = 0; t2 < 2; ++t2) {
    int tix = wave + t2 * 8;
    int row0 = (tix & 3) * 16, col = cb * 64 + (tix >> 2) * 16 + (lane & 15);
    float bias = b2v[col];
#pragma unroll
    for (int r = 0; r < 8; ++r) {
      size_t row = (size_t)rb * 64 + row0 + r + ((lane >> 4) << 3);
      out[row * 256 + col] = x1[row * 256 + col] + acc[t2][r] + bias;
    }
  }
}

extern "C" void kernel_launch(void* const* d_in, const int* in_sizes, int n_in,
                              void* d_out, int out_size, void* d_ws, size_t ws_size,
                              hipStream_t stream) {
  const float* x      = (const float*)d_in[0];
  const float* g1     = (const float*)d_in[1];
  const float* b1     = (const float*)d_in[2];
  const float* qkv_w  = (const float*)d_in[3];
  const float* qkv_b  = (const float*)d_in[4];
  const float* rpb    = (const float*)d_in[5];
  const float* proj_w = (const float*)d_in[6];
  const float* proj_b = (const float*)d_in[7];
  const float* g2     = (const float*)d_in[8];
  const float* b2     = (const float*)d_in[9];
  const float* fc1_w  = (const float*)d_in[10];
  const float* fc1_b  = (const float*)d_in[11];
  const float* fc2_w  = (const float*)d_in[12];
  const float* fc2_b  = (const float*)d_in[13];

  char* ws = (char*)d_ws;
  size_t off = 0;
  auto alloc = [&](size_t bytes) { char* p = ws + off; off += (bytes + 255) & ~(size_t)255; return p; };
  _Float16* Wq_h = (_Float16*)alloc((size_t)768 * 256 * 2);
  _Float16* Wp_h = (_Float16*)alloc((size_t)256 * 256 * 2);
  _Float16* W1_h = (_Float16*)alloc((size_t)1024 * 256 * 2);
  _Float16* W2_h = (_Float16*)alloc((size_t)256 * 1024 * 2);
  _Float16* hwin = (_Float16*)alloc((size_t)WINS * 64 * 256 * 2);
  _Float16* qbuf = (_Float16*)alloc((size_t)WINS * 8 * 64 * 32 * 2);
  _Float16* kbuf = (_Float16*)alloc((size_t)WINS * 8 * 64 * 32 * 2);
  _Float16* vbuf = (_Float16*)alloc((size_t)WINS * 8 * 64 * 32 * 2);
  _Float16* abuf = (_Float16*)alloc((size_t)WINS * 64 * 256 * 2);
  float*    x1   = (float*)   alloc((size_t)TOK * 256 * 4);
  _Float16* h2   = (_Float16*)alloc((size_t)TOK * 256 * 2);
  _Float16* h3   = (_Float16*)alloc((size_t)TOK * 1024 * 2);

  const int thr = 256;
  k_f32_to_f16<<<(768 * 256 + 255) / 256, thr, 0, stream>>>(qkv_w, Wq_h, 768 * 256);
  k_f32_to_f16<<<(256 * 256 + 255) / 256, thr, 0, stream>>>(proj_w, Wp_h, 256 * 256);
  k_f32_to_f16<<<(1024 * 256 + 255) / 256, thr, 0, stream>>>(fc1_w, W1_h, 1024 * 256);
  k_f32_to_f16<<<(256 * 1024 + 255) / 256, thr, 0, stream>>>(fc2_w, W2_h, 256 * 1024);

  k_ln1_shift<<<TOK / 8, thr, 0, stream>>>(x, g1, b1, hwin);
  k_qkv<<<WINS, thr, 0, stream>>>(hwin, Wq_h, qkv_b, qbuf, kbuf, vbuf);
  k_attn<<<WINS, thr, (size_t)8 * 64 * 72 * sizeof(_Float16), stream>>>(qbuf, kbuf, vbuf, rpb, abuf);
  k_proj<<<WINS, thr, 0, stream>>>(abuf, Wp_h, proj_b, x, x1);
  k_ln2<<<TOK / 8, thr, 0, stream>>>(x1, g2, b2, h2);
  k_fc1<<<dim3(TOK / 64, 16), thr, 0, stream>>>(h2, W1_h, fc1_b, h3);
  k_fc2<<<dim3(TOK / 64, 4), thr, 0, stream>>>(h3, W2_h, fc2_b, x1, (float*)d_out);
}
